// SingerHeadGatLayer_26328149524691
// MI455X (gfx1250) — matrix operand order repaired and checked by reading, not compiled, not run on hardware
//
#include <hip/hip_runtime.h>
#include <hip/hip_bf16.h>

// ---------------------------------------------------------------------------
// SingerHeadGatLayer on MI455X (gfx1250, wave32, WMMA).
// B=8, NA=NB=2048, in=256, out=64.
//   sA = h_A @ (WA_w @ aA_w) + (WA_b . aA_w + aA_b)     (collapsed projection)
//   z_B = h_B @ WB_w + WB_b ; sB = z_B @ aB_w + aB_b    (bf16 WMMA GEMM)
//   out = softmax_row(leaky(mask ? sA+sB : NEG)) @ z_B  (fused, WMMA)
// Softmax is computed WITHOUT running-max: alpha_i = exp(e_i)/sum_unmasked,
// which is exactly the reference result (masked terms contribute 0 to both
// numerator and denominator). Scores are pre-scaled by log2(e) so the exp is
// a single v_exp_f32. Sum-only state lets 4 waves split NB and merge via LDS.
// ---------------------------------------------------------------------------

typedef __attribute__((ext_vector_type(16))) __bf16 v16bf;
typedef __attribute__((ext_vector_type(8)))  float  v8f;

#define WMMA_BF16(A, Bm, C) \
  __builtin_amdgcn_wmma_f32_16x16x32_bf16(false, (A), false, (Bm), (short)0, (C), false, false)

static constexpr int Bn  = 8;
static constexpr int NA  = 2048;
static constexpr int NB  = 2048;
static constexpr int DIN = 256;
static constexpr int F   = 64;
static constexpr float LOG2E = 1.44269504088896340736f;

// ---------------------------------------------------------------------------
// Kernel 0: wa[k] = sum_f WA_w[k][f]*aA_w[f];  ca = WA_b.aA_w + aA_b;
//           WBt[f][k] = bf16(WB_w[k][f])   (B-operand friendly transpose)
// ---------------------------------------------------------------------------
__global__ void prep_kernel(const float* __restrict__ WAw, const float* __restrict__ WAb,
                            const float* __restrict__ aAw, const float* __restrict__ aAb,
                            const float* __restrict__ WBw,
                            float* __restrict__ wa, float* __restrict__ ca,
                            unsigned short* __restrict__ WBt_) {
  __bf16* WBt = (__bf16*)WBt_;
  int t = threadIdx.x;             // 256 threads
  float s = 0.f;
  for (int f = 0; f < F; ++f) s += WAw[t * F + f] * aAw[f];
  wa[t] = s;
  if (t == 0) {
    float c = aAb[0];
    for (int f = 0; f < F; ++f) c += WAb[f] * aAw[f];
    ca[0] = c;
  }
  for (int idx = t; idx < F * DIN; idx += 256) {
    int f = idx >> 8;              // out-dim
    int k = idx & (DIN - 1);       // in-dim
    WBt[idx] = (__bf16)WBw[k * F + f];
  }
}

// ---------------------------------------------------------------------------
// Kernel 1: sA[row] = (h_A[row].wa + ca) * log2(e)   (one row per wave)
// ---------------------------------------------------------------------------
__global__ void sA_kernel(const float* __restrict__ hA, const float* __restrict__ wa,
                          const float* __restrict__ ca, float* __restrict__ sA) {
  int wid  = (blockIdx.x * blockDim.x + threadIdx.x) >> 5;  // global wave = row
  int lane = threadIdx.x & 31;
  const float* hp = hA + (size_t)wid * DIN + lane * 8;
  const float* wp = wa + lane * 8;
  float s = 0.f;
#pragma unroll
  for (int i = 0; i < 8; ++i) s += hp[i] * wp[i];
#pragma unroll
  for (int off = 16; off; off >>= 1) s += __shfl_xor(s, off, 32);
  if (lane == 0) sA[wid] = (s + ca[0]) * LOG2E;
}

// ---------------------------------------------------------------------------
// Kernel 2: z_B projection with bf16 WMMA. One wave computes a 16x64 tile.
//           Stores z_B transposed bf16 [B][64][NB] and sB (log2-scaled).
// ---------------------------------------------------------------------------
__global__ void zB_kernel(const float* __restrict__ hB,
                          const unsigned short* __restrict__ WBt_,
                          const float* __restrict__ WBb,
                          const float* __restrict__ aBw, const float* __restrict__ aBb,
                          unsigned short* __restrict__ zBt_, float* __restrict__ sB) {
  const __bf16* WBt = (const __bf16*)WBt_;
  __bf16* zBt = (__bf16*)zBt_;

  int lane = threadIdx.x & 31;
  int tile = blockIdx.x * 4 + (threadIdx.x >> 5);   // 0..1023
  int b    = tile >> 7;                              // 128 tiles per batch
  int mrow = (tile & 127) << 4;
  int r15  = lane & 15;
  int hi   = (lane >> 4) & 1;                        // half-wave select
  int koff = hi ? 16 : 0;
  int hi8  = hi << 3;

  v8f acc0 = {}, acc1 = {}, acc2 = {}, acc3 = {};
  const float* hrow = hB + (size_t)(b * NB + mrow + r15) * DIN;

  for (int kc = 0; kc < DIN; kc += 32) {
    // A operand: 16x32 bf16 rows of h_B in contiguous k order (k = 16*half + s)
    v16bf a;
    const float* p0 = hrow + kc + koff;
    const float* p1 = hrow + kc + 8 + koff;
#pragma unroll
    for (int i = 0; i < 8; ++i) { a[i] = (__bf16)p0[i]; a[8 + i] = (__bf16)p1[i]; }
    int krun = kc + (hi ? 16 : 0);
    // B operands: contiguous 16xbf16 per lane from transposed weights
    v16bf b0 = *(const v16bf*)(WBt + (size_t)(r15)      * DIN + krun);
    v16bf b1 = *(const v16bf*)(WBt + (size_t)(16 + r15) * DIN + krun);
    v16bf b2 = *(const v16bf*)(WBt + (size_t)(32 + r15) * DIN + krun);
    v16bf b3 = *(const v16bf*)(WBt + (size_t)(48 + r15) * DIN + krun);
    acc0 = WMMA_BF16(a, b0, acc0);
    acc1 = WMMA_BF16(a, b1, acc1);
    acc2 = WMMA_BF16(a, b2, acc2);
    acc3 = WMMA_BF16(a, b3, acc3);
  }

  float bw0 = WBb[r15], bw1 = WBb[16 + r15], bw2 = WBb[32 + r15], bw3 = WBb[48 + r15];
  float aw0 = aBw[r15], aw1 = aBw[16 + r15], aw2 = aBw[32 + r15], aw3 = aBw[48 + r15];
  float abb = aBb[0];

#pragma unroll
  for (int r = 0; r < 8; ++r) {
    float v0 = acc0[r] + bw0, v1 = acc1[r] + bw1;
    float v2 = acc2[r] + bw2, v3 = acc3[r] + bw3;
    int row = mrow + r + hi8;
    zBt[(size_t)(b * F + r15)      * NB + row] = (__bf16)v0;
    zBt[(size_t)(b * F + 16 + r15) * NB + row] = (__bf16)v1;
    zBt[(size_t)(b * F + 32 + r15) * NB + row] = (__bf16)v2;
    zBt[(size_t)(b * F + 48 + r15) * NB + row] = (__bf16)v3;
    float p = v0 * aw0 + v1 * aw1 + v2 * aw2 + v3 * aw3;
    p += __shfl_xor(p, 1, 32);
    p += __shfl_xor(p, 2, 32);
    p += __shfl_xor(p, 4, 32);
    p += __shfl_xor(p, 8, 32);               // reduce within each 16-lane half
    if (r15 == r) sB[b * NB + row] = (p + abb) * LOG2E;
  }
}

// ---------------------------------------------------------------------------
// Kernel 3: fused GAT attention. One block (4 waves) owns a 16-row NA tile;
// each wave streams a 512-wide NB segment (chunks of 32), accumulating
// alpha-numerator @ z_B with WMMA; partial (acc, l) merged through LDS.
// ---------------------------------------------------------------------------
__global__ void attn_kernel(const float* __restrict__ sA, const float* __restrict__ sB,
                            const unsigned char* __restrict__ mask,
                            const unsigned short* __restrict__ zBt_,
                            float* __restrict__ out) {
  const __bf16* zb_all = (const __bf16*)zBt_;
  __shared__ float redA[4][16][F];   // per-wave 16x64 partial accumulators
  __shared__ float redL[4][16];      // per-wave row denominators

  int lane = threadIdx.x & 31;
  int w    = threadIdx.x >> 5;       // wave id 0..3 = NB segment
  int tile = blockIdx.x;             // 0..1023
  int b    = tile >> 7;
  int arow = (tile & 127) << 4;
  int r15  = lane & 15;
  int hi   = (lane >> 4) & 1;
  int koff = hi ? 16 : 0;
  int hi8  = hi << 3;

  int row = arow + r15;
  float sa = sA[b * NA + row];                       // log2-domain score half
  const unsigned char* mrow = mask + (size_t)(b * NA + row) * NB;
  const float* sbp = sB + b * NB;
  const __bf16* zb = zb_all + (size_t)b * F * NB;

  float l = 0.f;                                     // per-lane partial denom
  v8f acc0 = {}, acc1 = {}, acc2 = {}, acc3 = {};

  const int jb0 = w * (NB / 4);
  const int jb1 = jb0 + (NB / 4);
#pragma unroll 2
  for (int jb = jb0; jb < jb1; jb += 32) {
    __builtin_prefetch(mrow + jb + 256, 0, 1);       // global_prefetch_b8 ahead
    unsigned long long mk0 = *(const unsigned long long*)(mrow + jb + koff);
    unsigned long long mk1 = *(const unsigned long long*)(mrow + jb + 8 + koff);
    v8f sb0 = *(const v8f*)(sbp + jb + koff);
    v8f sb1 = *(const v8f*)(sbp + jb + 8 + koff);

    v16bf a;                                         // alpha numerators (bf16)
#pragma unroll
    for (int i = 0; i < 8; ++i) {
      float e0 = ((mk0 >> (8 * i)) & 0xFFull) ? (sa + sb0[i]) : -1.0e30f;
      float e1 = ((mk1 >> (8 * i)) & 0xFFull) ? (sa + sb1[i]) : -1.0e30f;
      e0 = e0 > 0.f ? e0 : 0.01f * e0;               // leaky (log2-homogeneous)
      e1 = e1 > 0.f ? e1 : 0.01f * e1;
      float p0 = __builtin_amdgcn_exp2f(fminf(e0, 115.f));  // masked -> 0
      float p1 = __builtin_amdgcn_exp2f(fminf(e1, 115.f));
      l += p0 + p1;
      a[i]     = (__bf16)p0;                         // already in A-matrix order
      a[8 + i] = (__bf16)p1;
    }

    int krun = jb + (hi ? 16 : 0);
    v16bf b0 = *(const v16bf*)(zb + (size_t)(r15)      * NB + krun);
    v16bf b1 = *(const v16bf*)(zb + (size_t)(16 + r15) * NB + krun);
    v16bf b2 = *(const v16bf*)(zb + (size_t)(32 + r15) * NB + krun);
    v16bf b3 = *(const v16bf*)(zb + (size_t)(48 + r15) * NB + krun);
    acc0 = WMMA_BF16(a, b0, acc0);
    acc1 = WMMA_BF16(a, b1, acc1);
    acc2 = WMMA_BF16(a, b2, acc2);
    acc3 = WMMA_BF16(a, b3, acc3);
  }

  // stash this wave's partials: halves of a row hold disjoint K-subsets of l
  float lsum = l + __shfl_xor(l, 16, 32);
  if (lane < 16) redL[w][lane] = lsum;
#pragma unroll
  for (int r = 0; r < 8; ++r) {
    int rr = r + hi8;                                // C-layout row of acc[r]
    redA[w][rr][r15]      = acc0[r];
    redA[w][rr][16 + r15] = acc1[r];
    redA[w][rr][32 + r15] = acc2[r];
    redA[w][rr][48 + r15] = acc3[r];
  }
  __syncthreads();

  // merge 4 segments and normalize: 128 threads, 8 outputs each
  int orow = threadIdx.x >> 3;                       // 0..15
  int ocol = (threadIdx.x & 7) * 8;                  // 0,8,...,56
  float lt = redL[0][orow] + redL[1][orow] + redL[2][orow] + redL[3][orow];
  float inv = lt > 0.f ? 1.f / lt : 0.f;             // fully-masked row -> 0
  size_t obase = (size_t)(b * NA + arow + orow) * F + ocol;
#pragma unroll
  for (int i = 0; i < 8; ++i) {
    float v = redA[0][orow][ocol + i] + redA[1][orow][ocol + i]
            + redA[2][orow][ocol + i] + redA[3][orow][ocol + i];
    out[obase + i] = v * inv;
  }
}

// ---------------------------------------------------------------------------
// Launch
// ---------------------------------------------------------------------------
extern "C" void kernel_launch(void* const* d_in, const int* in_sizes, int n_in,
                              void* d_out, int out_size, void* d_ws, size_t ws_size,
                              hipStream_t stream) {
  const float*         hA   = (const float*)d_in[0];
  const float*         hB   = (const float*)d_in[1];
  const unsigned char* mask = (const unsigned char*)d_in[2]; // jax bool = 1 byte
  const float*         WAw  = (const float*)d_in[3];
  const float*         WAb  = (const float*)d_in[4];
  const float*         WBw  = (const float*)d_in[5];
  const float*         WBb  = (const float*)d_in[6];
  const float*         aAw  = (const float*)d_in[7];
  const float*         aAb  = (const float*)d_in[8];
  const float*         aBw  = (const float*)d_in[9];
  const float*         aBb  = (const float*)d_in[10];
  float* out = (float*)d_out;

  char* ws = (char*)d_ws;
  float*          wa  = (float*)(ws + 0);              // 256 f32
  float*          ca  = (float*)(ws + 1024);           // 1 f32
  unsigned short* WBt = (unsigned short*)(ws + 1280);  // 64x256 bf16 (32 KB)
  float*          sA  = (float*)(ws + 34048);          // 8x2048 f32 (64 KB)
  float*          sB  = (float*)(ws + 99584);          // 8x2048 f32 (64 KB)
  unsigned short* zBt = (unsigned short*)(ws + 165120); // 8x64x2048 bf16 (2 MB)

  hipLaunchKernelGGL(prep_kernel, dim3(1), dim3(256), 0, stream,
                     WAw, WAb, aAw, aAb, WBw, wa, ca, WBt);
  hipLaunchKernelGGL(sA_kernel, dim3((Bn * NA) / 8), dim3(256), 0, stream,
                     hA, wa, ca, sA);
  hipLaunchKernelGGL(zB_kernel, dim3((Bn * NB / 16) / 4), dim3(128), 0, stream,
                     hB, WBt, WBb, aBw, aBb, zBt, sB);
  hipLaunchKernelGGL(attn_kernel, dim3(Bn * NA / 16), dim3(128), 0, stream,
                     sA, sB, mask, zBt, out);
}
